// Psum_mlp_58471684768370
// MI455X (gfx1250) — 8-bit matrix operand order repaired and checked by reading, compiled, not run on hardware
//
#include <hip/hip_runtime.h>
#include <hip/hip_bf16.h>

typedef int   v8i __attribute__((ext_vector_type(8)));
typedef float v8f __attribute__((ext_vector_type(8)));
typedef float v2f __attribute__((ext_vector_type(2)));

#define B_    1024
#define IN1   784
#define H_    512
#define NCLS  10

__global__ __launch_bounds__(32) void k_zero_scales(unsigned* a, unsigned* b) {
  if (threadIdx.x == 0) { *a = 0u; *b = 0u; }
}

__global__ __launch_bounds__(256) void k_absmax(const float* __restrict__ w, int n,
                                                unsigned* __restrict__ out) {
  __shared__ float sm[256];
  float m = 0.f;
  for (int i = blockIdx.x * blockDim.x + threadIdx.x; i < n; i += gridDim.x * blockDim.x)
    m = fmaxf(m, fabsf(w[i]));
  sm[threadIdx.x] = m;
  __syncthreads();
  for (int s = 128; s > 0; s >>= 1) {
    if (threadIdx.x < s) sm[threadIdx.x] = fmaxf(sm[threadIdx.x], sm[threadIdx.x + s]);
    __syncthreads();
  }
  if (threadIdx.x == 0) atomicMax(out, __float_as_uint(sm[0]));
}

__global__ __launch_bounds__(256) void k_wbits(const float* __restrict__ w,
                                               const unsigned* __restrict__ scbits,
                                               unsigned char* __restrict__ bf) {
  int idx = blockIdx.x * blockDim.x + threadIdx.x;
  int o = idx >> 9, k = idx & 511;
  float scale = __uint_as_float(*scbits) * (1.0f / 7.0f);
  float wq = rintf(fminf(fmaxf(w[idx] / scale, -7.f), 7.f));
  int wi = (int)wq;
  int mag0 = wi > 0 ? wi : 0;
  int mag1 = wi < 0 ? -wi : 0;
  int ntile = o >> 4, ni = o & 15, kstep = k >> 6, k64 = k & 63;
  int hi, v, byt;
  if (k64 < 32) { hi = k64 >> 4; v = (k64 & 15) >> 2; byt = k64 & 3; }
  else { int kk = k64 - 32; hi = kk >> 4; v = 4 + ((kk & 15) >> 2); byt = kk & 3; }
  int off = (hi * 16 + ni) * 32 + v * 4 + byt;
  int fb = (ntile * 8 + kstep) * 4;
  #pragma unroll
  for (int wb = 0; wb < 4; ++wb) {
    bf[(unsigned)((fb + wb) * 2 + 0) * 1024u + off] = (unsigned char)((mag0 >> wb) & 1);
    bf[(unsigned)((fb + wb) * 2 + 1) * 1024u + off] = (unsigned char)((mag1 >> wb) & 1);
  }
}

#if __has_builtin(__builtin_amdgcn_wmma_f32_16x16x4_f32)
#define HAVE_F32X4 1
#else
#define HAVE_F32X4 0
#endif

__global__ __launch_bounds__(256) void k_fc1(const float* __restrict__ x,
                                             const float* __restrict__ w1,
                                             const float* __restrict__ b1,
                                             float* __restrict__ h1) {
  int wave = threadIdx.x >> 5, lane = threadIdx.x & 31;
  int tile = blockIdx.x * 8 + wave;
  int mtile = tile >> 5, ntile = tile & 31;
  int m0 = mtile * 16, n0 = ntile * 16;
  int mi = lane & 15, hi = lane >> 4;
  v8f c = {0.f, 0.f, 0.f, 0.f, 0.f, 0.f, 0.f, 0.f};
#if HAVE_F32X4 && defined(__HIP_DEVICE_COMPILE__)
  const float* xr = x + (m0 + mi) * IN1 + 2 * hi;
  const float* wr = w1 + (n0 + mi) * IN1 + 2 * hi;
  for (int k0 = 0; k0 < IN1; k0 += 4) {
    v2f a, b;
    a.x = xr[k0]; a.y = xr[k0 + 1];
    b.x = wr[k0]; b.y = wr[k0 + 1];
    c = __builtin_amdgcn_wmma_f32_16x16x4_f32(false, a, false, b, (short)0, c, false, false);
  }
#else
  for (int r = 0; r < 8; ++r) {
    const float* xr = x + (m0 + r + 8 * hi) * IN1;
    const float* wr = w1 + (n0 + mi) * IN1;
    float s = 0.f;
    for (int k = 0; k < IN1; ++k) s = fmaf(xr[k], wr[k], s);
    c[r] = s;
  }
#endif
  int col = n0 + mi;
  float bias = b1[col];
  #pragma unroll
  for (int r = 0; r < 8; ++r)
    h1[(m0 + r + 8 * hi) * H_ + col] = c[r] + bias;
}

__global__ __launch_bounds__(256) void k_colstats(const float* __restrict__ h,
                                                  float* __restrict__ mean,
                                                  float* __restrict__ var,
                                                  int rows, int cols) {
  __shared__ float s1[256], s2[256];
  int col = blockIdx.x;
  float sum = 0.f, sq = 0.f;
  for (int r = threadIdx.x; r < rows; r += blockDim.x) {
    float v = h[r * cols + col];
    sum += v; sq += v * v;
  }
  s1[threadIdx.x] = sum; s2[threadIdx.x] = sq;
  __syncthreads();
  for (int s = 128; s > 0; s >>= 1) {
    if (threadIdx.x < s) { s1[threadIdx.x] += s1[threadIdx.x + s]; s2[threadIdx.x] += s2[threadIdx.x + s]; }
    __syncthreads();
  }
  if (threadIdx.x == 0) {
    float m = s1[0] / (float)rows;
    mean[col] = m;
    var[col] = s2[0] / (float)rows - m * m;
  }
}

__global__ __launch_bounds__(256) void k_qact_abits(const float* __restrict__ h,
    const float* __restrict__ mean, const float* __restrict__ var,
    const float* __restrict__ g, const float* __restrict__ b,
    unsigned char* __restrict__ af) {
  int idx = blockIdx.x * blockDim.x + threadIdx.x;
  int m = idx >> 9, k = idx & 511;
  float y = g[k] * (h[idx] - mean[k]) * rsqrtf(var[k] + 1e-5f) + b[k];
  y = fminf(fmaxf(y, 0.f), 1.f);
  int xi = (int)rintf(y * 15.0f);
  int mtile = m >> 4, mi = m & 15, kstep = k >> 6, k64 = k & 63;
  int pair = k64 >> 4, rem = k64 & 15;
  int hi = pair & 1, off = ((pair >> 1) << 2) + (rem >> 2), byt = k64 & 3;
  unsigned base = (unsigned)((mtile * 8 + kstep) * 4) * 1024u
                + (unsigned)((hi * 16 + mi) * 32 + off * 4 + byt);
  #pragma unroll
  for (int a = 0; a < 4; ++a)
    af[base + (unsigned)a * 1024u] = (unsigned char)((xi >> a) & 1);
}

__global__ __launch_bounds__(256) void k_psum(const unsigned char* __restrict__ af,
                                              const unsigned char* __restrict__ bf,
                                              const unsigned* __restrict__ scbits,
                                              float* __restrict__ hout) {
  int wave = threadIdx.x >> 5, lane = threadIdx.x & 31;
  int tile = blockIdx.x * 8 + wave;
  int mtile = tile >> 5, ntile = tile & 31;
  const unsigned char* ab = af + (unsigned)lane * 32u;
  const unsigned char* bb = bf + (unsigned)lane * 32u;
  v8i acc = {0, 0, 0, 0, 0, 0, 0, 0};
  for (int ch = 0; ch < 4; ++ch) {
    v8i A[4][2];
    #pragma unroll
    for (int a = 0; a < 4; ++a) {
      #pragma unroll
      for (int ks = 0; ks < 2; ++ks)
        A[a][ks] = *(const v8i*)(ab + (unsigned)((mtile * 8 + ch * 2 + ks) * 4 + a) * 1024u);
    }
    #pragma unroll
    for (int wb = 0; wb < 4; ++wb) {
      #pragma unroll
      for (int s = 0; s < 2; ++s) {
        v8i B0 = *(const v8i*)(bb + (unsigned)((((ntile * 8 + ch * 2 + 0) * 4 + wb) * 2) + s) * 1024u);
        v8i B1 = *(const v8i*)(bb + (unsigned)((((ntile * 8 + ch * 2 + 1) * 4 + wb) * 2) + s) * 1024u);
        #pragma unroll
        for (int a = 0; a < 4; ++a) {
          v8i ps = {0, 0, 0, 0, 0, 0, 0, 0};
          ps = __builtin_amdgcn_wmma_i32_16x16x64_iu8(false, A[a][0], false, B0, ps, false, false);
          ps = __builtin_amdgcn_wmma_i32_16x16x64_iu8(false, A[a][1], false, B1, ps, false, false);
          int sh = a + wb;
          #pragma unroll
          for (int r = 0; r < 8; ++r) {
            int q = (15 * ps[r] + 63) >> 7;
            acc[r] += s ? -(q << sh) : (q << sh);
          }
        }
      }
    }
  }
  float scale = __uint_as_float(*scbits) * (1.0f / 7.0f);
  float fs = scale * (128.0f / 225.0f);
  int mi = lane & 15, hi = lane >> 4;
  int col = ntile * 16 + mi;
  #pragma unroll
  for (int r = 0; r < 8; ++r)
    hout[(mtile * 16 + r + 8 * hi) * H_ + col] = (float)acc[r] * fs;
}

__global__ __launch_bounds__(256) void k_bn_apply(const float* __restrict__ h,
    const float* __restrict__ mean, const float* __restrict__ var,
    const float* __restrict__ g, const float* __restrict__ b,
    float* __restrict__ o, int n, int cols, int dorelu) {
  int idx = blockIdx.x * blockDim.x + threadIdx.x;
  if (idx >= n) return;
  int c = idx % cols;
  float y = g[c] * (h[idx] - mean[c]) * rsqrtf(var[c] + 1e-5f) + b[c];
  if (dorelu) y = fmaxf(y, 0.f);
  o[idx] = y;
}

__global__ __launch_bounds__(256) void k_fc4(const float* __restrict__ h,
                                             const float* __restrict__ w,
                                             const float* __restrict__ bias,
                                             float* __restrict__ t) {
  int idx = blockIdx.x * blockDim.x + threadIdx.x;
  if (idx >= B_ * NCLS) return;
  int bb = idx / NCLS, c = idx % NCLS;
  const float* hr = h + bb * H_;
  const float* wr = w + c * H_;
  float s = bias[c];
  for (int k = 0; k < H_; ++k) s = fmaf(hr[k], wr[k], s);
  t[idx] = s;
}

extern "C" void kernel_launch(void* const* d_in, const int* in_sizes, int n_in,
                              void* d_out, int out_size, void* d_ws, size_t ws_size,
                              hipStream_t stream) {
  const float* x   = (const float*)d_in[0];
  const float* w1  = (const float*)d_in[1];
  const float* b1  = (const float*)d_in[2];
  const float* g1  = (const float*)d_in[3];
  const float* bb1 = (const float*)d_in[4];
  const float* w2  = (const float*)d_in[5];
  const float* g2  = (const float*)d_in[6];
  const float* bb2 = (const float*)d_in[7];
  const float* w3  = (const float*)d_in[8];
  const float* g3  = (const float*)d_in[9];
  const float* bb3 = (const float*)d_in[10];
  const float* w4  = (const float*)d_in[11];
  const float* b4  = (const float*)d_in[12];
  const float* g4  = (const float*)d_in[13];
  const float* bb4 = (const float*)d_in[14];
  float* out = (float*)d_out;

  char* ws = (char*)d_ws;
  const size_t MB = 1u << 20;
  float* H1 = (float*)(ws + 0 * MB);
  float* H2 = (float*)(ws + 2 * MB);
  float* H3 = (float*)(ws + 4 * MB);
  unsigned char* AF  = (unsigned char*)(ws + 6 * MB);
  unsigned char* BF2 = (unsigned char*)(ws + 8 * MB);
  unsigned char* BF3 = (unsigned char*)(ws + 10 * MB);
  char* ST = ws + 12 * MB;
  float*    mean  = (float*)(ST);
  float*    var   = (float*)(ST + 2048);
  unsigned* sc2   = (unsigned*)(ST + 4096);
  unsigned* sc3   = (unsigned*)(ST + 4100);
  float*    mean4 = (float*)(ST + 4104);
  float*    var4  = (float*)(ST + 4160);
  float*    T4    = (float*)(ST + 4224);
  float*    H3R   = H1;

  k_zero_scales<<<1, 32, 0, stream>>>(sc2, sc3);
  k_absmax<<<256, 256, 0, stream>>>(w2, H_ * H_, sc2);
  k_absmax<<<256, 256, 0, stream>>>(w3, H_ * H_, sc3);
  k_wbits<<<(H_ * H_) / 256, 256, 0, stream>>>(w2, sc2, BF2);
  k_wbits<<<(H_ * H_) / 256, 256, 0, stream>>>(w3, sc3, BF3);

  k_fc1<<<256, 256, 0, stream>>>(x, w1, b1, H1);
  k_colstats<<<H_, 256, 0, stream>>>(H1, mean, var, B_, H_);
  k_qact_abits<<<(B_ * H_) / 256, 256, 0, stream>>>(H1, mean, var, g1, bb1, AF);

  k_psum<<<256, 256, 0, stream>>>(AF, BF2, sc2, H2);
  k_colstats<<<H_, 256, 0, stream>>>(H2, mean, var, B_, H_);
  k_qact_abits<<<(B_ * H_) / 256, 256, 0, stream>>>(H2, mean, var, g2, bb2, AF);

  k_psum<<<256, 256, 0, stream>>>(AF, BF3, sc3, H3);
  k_colstats<<<H_, 256, 0, stream>>>(H3, mean, var, B_, H_);
  k_bn_apply<<<(B_ * H_) / 256, 256, 0, stream>>>(H3, mean, var, g3, bb3, H3R, B_ * H_, H_, 1);

  k_fc4<<<(B_ * NCLS + 255) / 256, 256, 0, stream>>>(H3R, w4, b4, T4);
  k_colstats<<<NCLS, 256, 0, stream>>>(T4, mean4, var4, B_, NCLS);
  k_bn_apply<<<(B_ * NCLS + 255) / 256, 256, 0, stream>>>(T4, mean4, var4, g4, bb4, out, B_ * NCLS, NCLS, 0);
}
